// S4_77833397338699
// MI455X (gfx1250) — compile-verified, matrix operand order checked by reading, not run on hardware
//
#include <hip/hip_runtime.h>
#include <math.h>
#include <stdint.h>

// ---------------- problem constants ----------------
#define BATCH 8
#define SEQ   2048
#define DM    512      // d_model == H
#define DF    2048     // d_ffn
#define NST   64       // ssm states
#define KSZ   31       // depthwise kernel
#define NL    6
#define MROWS (BATCH*SEQ)   // 16384 rows
#define KSRC  80
#define KSRCP 96            // padded to x32

typedef __attribute__((ext_vector_type(16))) __bf16 v16bf;
typedef __attribute__((ext_vector_type(8)))  __bf16 v8bf;
typedef __attribute__((ext_vector_type(8)))  float  v8f;
typedef __attribute__((ext_vector_type(4)))  unsigned int v4u;
typedef __attribute__((ext_vector_type(8)))  int v8i;
typedef __attribute__((ext_vector_type(4)))  int v4i;

// ---------------- CDNA5 feature probes (compile-safe) ----------------
#ifndef __has_builtin
#define __has_builtin(x) 0
#endif
#if __has_builtin(__builtin_amdgcn_tensor_load_to_lds) && \
    __has_builtin(__builtin_amdgcn_s_wait_tensorcnt)
#define USE_TDM 1
#else
#define USE_TDM 0
#endif
#if __has_builtin(__builtin_amdgcn_global_load_async_to_lds_b128) && \
    __has_builtin(__builtin_amdgcn_s_wait_asynccnt)
#define USE_ASYNC 1
#else
#define USE_ASYNC 0
#endif

typedef __attribute__((address_space(3))) void lds_void_t;
typedef __attribute__((address_space(1))) v4i  g_v4i;   // global int4*
typedef __attribute__((address_space(3))) v4i  l_v4i;   // LDS int4*

__device__ __forceinline__ unsigned lds_addr_of(void* p) {
  return (unsigned)(uintptr_t)(lds_void_t*)p;   // addrspacecast generic->LDS, then 32-bit offset
}

__device__ __forceinline__ float gelu_f(float x) {
  float x3 = x * x * x;
  return 0.5f * x * (1.0f + tanhf(0.7978845608028654f * (x + 0.044715f * x3)));
}
__device__ __forceinline__ float sigmoid_f(float x) {
  return 1.0f / (1.0f + expf(-x));
}

// =====================================================================
// Weight pre-pack: Wb[n*Kp + k] = bf16( trans ? W[n*K+k] : W[k*N+n] ),
// zero-padded for k >= K. Run once per call; weights reused M=16384x.
// =====================================================================
__global__ __launch_bounds__(256)
void pack_w_kernel(const float* __restrict__ W, __bf16* __restrict__ Wb,
                   int N, int K, int Kp, int trans, int tot)
{
  const int idx = blockIdx.x * blockDim.x + threadIdx.x;
  if (idx >= tot) return;
  const int n = idx / Kp, k = idx - n * Kp;
  float v = 0.f;
  if (k < K) v = trans ? W[(size_t)n * K + k] : W[(size_t)k * N + n];
  Wb[idx] = (__bf16)v;
}

// Zero-pad activations along K (only needed for the 80-wide input proj)
__global__ __launch_bounds__(256)
void pad_a_kernel(const float* __restrict__ src, float* __restrict__ As,
                  int K, int Kp, int tot)
{
  const int idx = blockIdx.x * blockDim.x + threadIdx.x;
  if (idx >= tot) return;
  const int m = idx / Kp, k = idx - m * Kp;
  As[idx] = (k < K) ? src[(size_t)m * K + k] : 0.f;
}

// =====================================================================
// GEMM: C[M,N] = (res? res : 0) + alpha * act( A[M,lda fp32] @ Wb^T + bias )
// Wb is packed bf16 [N, Kp] row-major (Kp % 32 == 0).
// Block tile 128(M) x 64(N), 8 waves; wave = 2x2 WMMA 16x16x32 bf16 tiles.
// A tile staged fp32 -> LDS via async-load-to-LDS; B tile staged via TDM.
// =====================================================================
#define GEMM_BM 128
#define GEMM_BN 64

__global__ __launch_bounds__(256)
void gemm_bf16_wmma(const float* __restrict__ A, int lda,
                    const __bf16* __restrict__ Wb, int Kp,
                    const float* __restrict__ bias, const float* res,
                    float* C, int N, int doGelu, float alpha)
{
  __shared__ __align__(16) float  As[GEMM_BM][32];   // 16KB fp32 A tile
  __shared__ __align__(16) __bf16 Bs[GEMM_BN][32];   //  4KB bf16 B tile

  const int tid   = threadIdx.x;
  const int lane  = tid & 31;
  const int wave  = tid >> 5;
  const int waveM = wave & 3;       // 0..3  -> 32-row slabs
  const int waveN = wave >> 2;      // 0..1  -> 32-col slabs
  const int blockM = blockIdx.y * GEMM_BM;
  const int blockN = blockIdx.x * GEMM_BN;
  const int lhalf = lane >> 4;      // 0 or 1
  const int l15   = lane & 15;
  const int kg    = lhalf << 3;     // K-groups {0..7,16..23} / {8..15,24..31}

  v8f acc[2][2];
  const v8f vzero = {0.f,0.f,0.f,0.f,0.f,0.f,0.f,0.f};
  for (int i = 0; i < 2; ++i)
    for (int j = 0; j < 2; ++j)
      acc[i][j] = vzero;

  const int a_m = tid >> 1;           // 0..127 (row of A tile)
  const int a_c = (tid & 1) * 16;     // fp32 column 0 or 16
  const int b_n = tid >> 2;           // 0..63  (row of B tile)
  const int b_c = (tid & 3) * 8;      // bf16 column chunk

  for (int k0 = 0; k0 < Kp; k0 += 32) {
    // ---------------- stage A tile (fp32, 128x32) ----------------
    {
      const float* gp = A + (size_t)(blockM + a_m) * lda + k0 + a_c;
      __builtin_prefetch(gp + 32, 0, 1);          // next K tile
#if USE_ASYNC
      #pragma unroll
      for (int c = 0; c < 4; ++c) {
        __builtin_amdgcn_global_load_async_to_lds_b128(
            (g_v4i*)(uintptr_t)(gp + c * 4),
            (l_v4i*)(v4i*)(void*)&As[a_m][a_c + c * 4],
            0, 0);
      }
#else
      float4* dst = (float4*)&As[a_m][a_c];
      #pragma unroll
      for (int c = 0; c < 4; ++c) dst[c] = *(const float4*)(gp + c * 4);
#endif
    }

    // ---------------- stage B tile (bf16, 64x32) ----------------
#if USE_TDM
    if (wave == 0) {
      const uint64_t ga = (uint64_t)(uintptr_t)(Wb + (size_t)blockN * Kp + k0);
      const unsigned la = lds_addr_of(&Bs[0][0]);
      // D# group0: count=1 | lds_addr | global_addr[56:0] | type=2
      v4u g0 = { 1u, la, (unsigned)ga,
                 (unsigned)((ga >> 32) & 0x1FFFFFFu) | (2u << 30) };
      // D# group1: data_size=2B(code 1); tile 32(k) x 64(n); stride0=Kp
      const unsigned td0 = (unsigned)Kp;     // tensor dim0 (elements)
      const unsigned td1 = (unsigned)N;      // tensor dim1 (rows)
      v8i g1 = { (int)(1u << 16),
                 (int)((td0 & 0xFFFFu) << 16),
                 (int)((td0 >> 16) | ((td1 & 0xFFFFu) << 16)),
                 (int)((td1 >> 16) | (32u << 16)),            // tile_dim0 = 32
                 (int)64u,                                     // tile_dim1 = 64
                 (int)(unsigned)Kp,                            // stride0 lo
                 0, 0 };                                       // stride0 hi, stride1
      v4i gz = {0,0,0,0};
#if __clang_major__ >= 23
      v8i gz8 = {0,0,0,0,0,0,0,0};
      __builtin_amdgcn_tensor_load_to_lds(g0, g1, gz, gz, gz8, 0);
#else
      __builtin_amdgcn_tensor_load_to_lds(g0, g1, gz, gz, 0);
#endif
      __builtin_amdgcn_s_wait_tensorcnt(0);
    }
#else
    *(uint4*)&Bs[b_n][b_c] =
        *(const uint4*)(Wb + (size_t)(blockN + b_n) * Kp + k0 + b_c);
#endif

#if USE_ASYNC
    __builtin_amdgcn_s_wait_asynccnt(0);
#endif
    __syncthreads();

    // ---- A fragments: LDS fp32 -> bf16 (layout: lane<16 K{0-7,16-23}) ----
    v16bf afrag[2];
    #pragma unroll
    for (int mi = 0; mi < 2; ++mi) {
      const float* ap = &As[waveM*32 + mi*16 + l15][kg];
      float4 f0 = *(const float4*)(ap);
      float4 f1 = *(const float4*)(ap + 4);
      float4 f2 = *(const float4*)(ap + 16);
      float4 f3 = *(const float4*)(ap + 20);
      v16bf af;
      af[0]=(__bf16)f0.x;  af[1]=(__bf16)f0.y;  af[2]=(__bf16)f0.z;  af[3]=(__bf16)f0.w;
      af[4]=(__bf16)f1.x;  af[5]=(__bf16)f1.y;  af[6]=(__bf16)f1.z;  af[7]=(__bf16)f1.w;
      af[8]=(__bf16)f2.x;  af[9]=(__bf16)f2.y;  af[10]=(__bf16)f2.z; af[11]=(__bf16)f2.w;
      af[12]=(__bf16)f3.x; af[13]=(__bf16)f3.y; af[14]=(__bf16)f3.z; af[15]=(__bf16)f3.w;
      afrag[mi] = af;
    }

    // ---- B fragments: two 128-bit ds loads per 16x32 tile ----
    v16bf bfrag[2];
    #pragma unroll
    for (int ni = 0; ni < 2; ++ni) {
      const __bf16* bp = &Bs[waveN*32 + ni*16 + l15][kg];
      v8bf b0 = *(const v8bf*)bp;
      v8bf b1 = *(const v8bf*)(bp + 16);
      v16bf bf;
      #pragma unroll
      for (int e = 0; e < 8; ++e) { bf[e] = b0[e]; bf[8+e] = b1[e]; }
      bfrag[ni] = bf;
    }

    // ---- 4 WMMAs ----
    #pragma unroll
    for (int mi = 0; mi < 2; ++mi)
      #pragma unroll
      for (int ni = 0; ni < 2; ++ni)
        acc[mi][ni] = __builtin_amdgcn_wmma_f32_16x16x32_bf16(
            false, afrag[mi], false, bfrag[ni],
            (short)0, acc[mi][ni], false, false);

    __syncthreads();
  }

  // ---- epilogue: lane l -> col = l&15; VGPR r -> row = r + 8*(l>=16) ----
  #pragma unroll
  for (int mi = 0; mi < 2; ++mi) {
    #pragma unroll
    for (int ni = 0; ni < 2; ++ni) {
      const int col = blockN + waveN*32 + ni*16 + l15;
      const float bv = bias[col];
      #pragma unroll
      for (int r = 0; r < 8; ++r) {
        const int row = blockM + waveM*32 + mi*16 + lhalf*8 + r;
        float v = acc[mi][ni][r] + bv;
        if (doGelu) v = gelu_f(v);
        v *= alpha;
        const size_t off = (size_t)row * N + col;
        if (res) v += res[off];
        C[off] = v;
      }
    }
  }
}

// =====================================================================
// LayerNorm over last dim (D=512), optional fused GELU / speaker add.
// =====================================================================
__global__ __launch_bounds__(256)
void ln_kernel(const float* __restrict__ in, float* out,
               const float* __restrict__ gg, const float* __restrict__ bb,
               const float* __restrict__ speaker,
               int T, int D, float eps, int doGelu)
{
  const int row  = blockIdx.x;
  const int tid  = threadIdx.x;
  const int lane = tid & 31;
  const int wave = tid >> 5;
  const float* rp = in + (size_t)row * D;
  float v0 = rp[tid];
  float v1 = rp[tid + 256];
  float s = v0 + v1;
  float q = v0*v0 + v1*v1;
  #pragma unroll
  for (int m = 16; m > 0; m >>= 1) {
    s += __shfl_xor(s, m, 32);
    q += __shfl_xor(q, m, 32);
  }
  __shared__ float red[16];
  if (lane == 0) { red[wave] = s; red[8 + wave] = q; }
  __syncthreads();
  if (tid == 0) {
    float ts = 0.f, tq = 0.f;
    for (int w = 0; w < 8; ++w) { ts += red[w]; tq += red[8 + w]; }
    float mean = ts / (float)D;
    float var  = tq / (float)D - mean * mean;
    red[0] = mean;
    red[8] = rsqrtf(var + eps);
  }
  __syncthreads();
  const float mean = red[0], rstd = red[8];
  float* op = out + (size_t)row * D;
  const float* spk = speaker ? (speaker + (size_t)(row / T) * D) : nullptr;
  {
    float y = (v0 - mean) * rstd * gg[tid] + bb[tid];
    if (doGelu) y = gelu_f(y);
    if (spk) y += spk[tid];
    op[tid] = y;
  }
  {
    const int d = tid + 256;
    float y = (v1 - mean) * rstd * gg[d] + bb[d];
    if (doGelu) y = gelu_f(y);
    if (spk) y += spk[d];
    op[d] = y;
  }
}

// =====================================================================
// DSS diagonal state-space scan (exact recurrent form of the FFT conv):
//   A_n = exp(lam_n*dt); S_n[t] = A_n*S_n[t-1] + u[t]; y = dt*Re(sum C_n S_n)
//   out = gelu(y + d_skip*u), [B,T,H] layout. One wave32 per (b,h).
// =====================================================================
__global__ __launch_bounds__(32)
void dss_scan(const float* __restrict__ xn, float* __restrict__ out,
              const float* __restrict__ lam_re, const float* __restrict__ lam_im,
              const float* __restrict__ c_re, const float* __restrict__ c_im,
              const float* __restrict__ d_skip, const float* __restrict__ log_dt)
{
  const int b = blockIdx.x / DM;
  const int h = blockIdx.x % DM;
  const int lane = threadIdx.x;
  const float dt = expf(log_dt[0]);

  float a0r, a0i, a1r, a1i;
  {
    float lr = lam_re[lane], li = lam_im[lane];
    float dec = expf(-expf(lr) * dt), ang = li * dt;
    a0r = dec * cosf(ang); a0i = dec * sinf(ang);
  }
  {
    float lr = lam_re[lane + 32], li = lam_im[lane + 32];
    float dec = expf(-expf(lr) * dt), ang = li * dt;
    a1r = dec * cosf(ang); a1i = dec * sinf(ang);
  }
  const float c0r = c_re[h*NST + lane],      c0i = c_im[h*NST + lane];
  const float c1r = c_re[h*NST + lane + 32], c1i = c_im[h*NST + lane + 32];
  const float dsk = d_skip[h];

  float s0r = 0.f, s0i = 0.f, s1r = 0.f, s1i = 0.f;
  const size_t base = ((size_t)b * SEQ) * DM + h;

  for (int t0 = 0; t0 < SEQ; t0 += 32) {
    const float u = xn[base + (size_t)(t0 + lane) * DM];
    float yout = 0.f;
    for (int j = 0; j < 32; ++j) {
      const float uj = __shfl(u, j, 32);
      float tr = a0r*s0r - a0i*s0i + uj;
      float ti = a0r*s0i + a0i*s0r;
      s0r = tr; s0i = ti;
      tr = a1r*s1r - a1i*s1i + uj;
      ti = a1r*s1i + a1i*s1r;
      s1r = tr; s1i = ti;
      float contrib = c0r*s0r - c0i*s0i + c1r*s1r - c1i*s1i;
      #pragma unroll
      for (int m = 16; m > 0; m >>= 1) contrib += __shfl_xor(contrib, m, 32);
      if (lane == j) yout = contrib;
    }
    const float y = dt * yout + dsk * u;
    out[base + (size_t)(t0 + lane) * DM] = gelu_f(y);
  }
}

// =====================================================================
// GLU: out[m,d] = (res? res[m,d]:0) + z[m,d]*sigmoid(z[m,D+d])
// =====================================================================
__global__ __launch_bounds__(256)
void glu_kernel(const float* __restrict__ z, const float* res,
                float* out, int tot, int D)
{
  const int idx = blockIdx.x * blockDim.x + threadIdx.x;
  if (idx >= tot) return;
  const int m = idx / D, d = idx - m * D;
  const size_t zo = (size_t)m * (2 * D);
  float v = z[zo + d] * sigmoid_f(z[zo + D + d]);
  if (res) v += res[idx];
  out[idx] = v;
}

// =====================================================================
// Depthwise conv, SAME padding, k=31, layout [B,T,D] (coalesced over d)
// =====================================================================
__global__ __launch_bounds__(256)
void dw_kernel(const float* __restrict__ in, const float* __restrict__ w,
               const float* __restrict__ bias, float* __restrict__ out, int tot)
{
  const int idx = blockIdx.x * blockDim.x + threadIdx.x;
  if (idx >= tot) return;
  const int d = idx % DM;
  const int t = (idx / DM) % SEQ;
  const int b = idx / (DM * SEQ);
  float acc = bias[d];
  const float* ip = in + ((size_t)b * SEQ) * DM + d;
  const float* wp = w + d * KSZ;
  #pragma unroll
  for (int j = 0; j < KSZ; ++j) {
    const int tt = t + j - 15;
    if (tt >= 0 && tt < SEQ) acc += ip[(size_t)tt * DM] * wp[j];
  }
  out[idx] = acc;
}

// =====================================================================
// Host orchestration
// =====================================================================
extern "C" void kernel_launch(void* const* d_in, const int* in_sizes, int n_in,
                              void* d_out, int out_size, void* d_ws, size_t ws_size,
                              hipStream_t stream) {
  const float* src        = (const float*)d_in[0];
  const float* speaker    = (const float*)d_in[1];
  const float* src_w      = (const float*)d_in[2];
  const float* src_b      = (const float*)d_in[3];
  const float* ffn1_ln_g  = (const float*)d_in[4];
  const float* ffn1_ln_b  = (const float*)d_in[5];
  const float* ffn1_w1    = (const float*)d_in[6];
  const float* ffn1_b1    = (const float*)d_in[7];
  const float* ffn1_w2    = (const float*)d_in[8];
  const float* ffn1_b2    = (const float*)d_in[9];
  const float* norm1_g    = (const float*)d_in[10];
  const float* norm1_b    = (const float*)d_in[11];
  const float* lam_re     = (const float*)d_in[12];
  const float* lam_im     = (const float*)d_in[13];
  const float* c_re       = (const float*)d_in[14];
  const float* c_im       = (const float*)d_in[15];
  const float* ssm_d      = (const float*)d_in[16];
  const float* log_dt     = (const float*)d_in[17];
  const float* dss_out_w  = (const float*)d_in[18];
  const float* dss_out_b  = (const float*)d_in[19];
  const float* conv_ln_g  = (const float*)d_in[20];
  const float* conv_ln_b  = (const float*)d_in[21];
  const float* bneck_w    = (const float*)d_in[22];
  const float* bneck_b    = (const float*)d_in[23];
  const float* dw_w       = (const float*)d_in[24];
  const float* dw_b       = (const float*)d_in[25];
  const float* conv_ln2_g = (const float*)d_in[26];
  const float* conv_ln2_b = (const float*)d_in[27];
  const float* conv_lin_w = (const float*)d_in[28];
  const float* conv_lin_b = (const float*)d_in[29];
  const float* ffn2_ln_g  = (const float*)d_in[30];
  const float* ffn2_ln_b  = (const float*)d_in[31];
  const float* ffn2_w1    = (const float*)d_in[32];
  const float* ffn2_b1    = (const float*)d_in[33];
  const float* ffn2_w2    = (const float*)d_in[34];
  const float* ffn2_b2    = (const float*)d_in[35];
  const float* norm2_g    = (const float*)d_in[36];
  const float* norm2_b    = (const float*)d_in[37];
  const float* final_g    = (const float*)d_in[38];
  const float* final_b    = (const float*)d_in[39];

  const int M = MROWS;
  const size_t MD = (size_t)M * DM;

  // ---- fp32 workspace ----
  float* X   = (float*)d_ws;           // [M, D]   residual stream
  float* XN  = X  + MD;                // [M, D]
  float* G1  = XN + MD;                // [M, D]
  float* Y2  = G1 + MD;                // [M, 2D]
  float* H1  = Y2 + 2 * MD;            // [M, F]
  float* Asp = H1 + 4 * MD;            // [M, 96]  padded src
  // ---- packed bf16 weights ----
  __bf16* WP = (__bf16*)(Asp + (size_t)M * KSRCP);
  size_t o = 0;
  __bf16* p_src = WP + o;  o += (size_t)DM * KSRCP;
  __bf16* p_w1  = WP + o;  o += (size_t)NL * DF * DM;   // [F, D]
  __bf16* p_w2  = WP + o;  o += (size_t)NL * DM * DF;   // [D, F]
  __bf16* p_dss = WP + o;  o += (size_t)NL * 2*DM * DM; // [2D, D]
  __bf16* p_bk  = WP + o;  o += (size_t)NL * 2*DM * DM; // [2D, D]
  __bf16* p_cl  = WP + o;  o += (size_t)NL * DM * DM;   // [D, D]
  __bf16* p_f21 = WP + o;  o += (size_t)NL * DF * DM;   // [F, D]
  __bf16* p_f22 = WP + o;  o += (size_t)NL * DM * DF;   // [D, F]

  auto pack = [&](const float* W, __bf16* Wb, int N, int K, int Kp, int trans) {
    const int tot = N * Kp;
    pack_w_kernel<<<(tot + 255) / 256, 256, 0, stream>>>(W, Wb, N, K, Kp, trans, tot);
  };
  auto gemm = [&](const float* A, int lda, const __bf16* Wb, int Kp,
                  const float* bias, const float* res, float* C,
                  int N, int act, float alpha) {
    dim3 grid(N / GEMM_BN, M / GEMM_BM);
    gemm_bf16_wmma<<<grid, 256, 0, stream>>>(A, lda, Wb, Kp, bias, res, C,
                                             N, act, alpha);
  };
  auto ln = [&](const float* in, float* out, const float* g, const float* b,
                const float* spk, float eps, int doGelu) {
    ln_kernel<<<M, 256, 0, stream>>>(in, out, g, b, spk, SEQ, DM, eps, doGelu);
  };

  // ---- one-time per call: pack weights to bf16 [N,Kp], pad src ----
  pack(src_w, p_src, DM, KSRC, KSRCP, 0);
  for (int i = 0; i < NL; ++i) {
    pack(ffn1_w1    + (size_t)i*DM*DF,   p_w1  + (size_t)i*DF*DM,   DF,   DM, DM, 0);
    pack(ffn1_w2    + (size_t)i*DF*DM,   p_w2  + (size_t)i*DM*DF,   DM,   DF, DF, 0);
    pack(dss_out_w  + (size_t)i*DM*2*DM, p_dss + (size_t)i*2*DM*DM, 2*DM, DM, DM, 0);
    pack(bneck_w    + (size_t)i*2*DM*DM, p_bk  + (size_t)i*2*DM*DM, 2*DM, DM, DM, 1);
    pack(conv_lin_w + (size_t)i*DM*DM,   p_cl  + (size_t)i*DM*DM,   DM,   DM, DM, 1);
    pack(ffn2_w1    + (size_t)i*DM*DF,   p_f21 + (size_t)i*DF*DM,   DF,   DM, DM, 0);
    pack(ffn2_w2    + (size_t)i*DF*DM,   p_f22 + (size_t)i*DM*DF,   DM,   DF, DF, 0);
  }
  {
    const int tot = M * KSRCP;
    pad_a_kernel<<<(tot + 255) / 256, 256, 0, stream>>>(src, Asp, KSRC, KSRCP, tot);
  }

  const int totD = M * DM;
  const int nbD  = (totD + 255) / 256;

  // ---- input projection ----
  gemm(Asp, KSRCP, p_src, KSRCP, src_b, nullptr, X, DM, 0, 1.0f);

  for (int i = 0; i < NL; ++i) {
    const size_t iD = (size_t)i * DM;
    // FFN1: X += 0.5 * (gelu(LN(X) @ w1 + b1) @ w2 + b2)
    ln(X, XN, ffn1_ln_g + iD, ffn1_ln_b + iD, nullptr, 1e-5f, 0);
    gemm(XN, DM, p_w1 + (size_t)i*DF*DM, DM, ffn1_b1 + (size_t)i*DF,
         nullptr, H1, DF, 1, 1.0f);
    gemm(H1, DF, p_w2 + (size_t)i*DM*DF, DF, ffn1_b2 + iD, X, X, DM, 0, 0.5f);

    // DSS: XN = LN(X); G1 = gelu(scan(XN)); X += GLU(G1 @ w_out + b)
    ln(X, XN, norm1_g + iD, norm1_b + iD, nullptr, 1e-5f, 0);
    dss_scan<<<BATCH * DM, 32, 0, stream>>>(
        XN, G1, lam_re + (size_t)i*NST, lam_im + (size_t)i*NST,
        c_re + (size_t)i*DM*NST, c_im + (size_t)i*DM*NST,
        ssm_d + iD, log_dt + i);
    gemm(G1, DM, p_dss + (size_t)i*2*DM*DM, DM, dss_out_b + (size_t)i*2*DM,
         nullptr, Y2, 2*DM, 0, 1.0f);
    glu_kernel<<<nbD, 256, 0, stream>>>(Y2, X, X, totD, DM);

    // Conv module: X += lin(gelu(LN2(dwconv(GLU(LN(X)@bw^T + bb)))))
    ln(X, XN, conv_ln_g + iD, conv_ln_b + iD, nullptr, 1e-5f, 0);
    gemm(XN, DM, p_bk + (size_t)i*2*DM*DM, DM, bneck_b + (size_t)i*2*DM,
         nullptr, Y2, 2*DM, 0, 1.0f);
    glu_kernel<<<nbD, 256, 0, stream>>>(Y2, nullptr, G1, totD, DM);
    dw_kernel<<<nbD, 256, 0, stream>>>(G1, dw_w + (size_t)i*DM*KSZ,
                                       dw_b + iD, XN, totD);
    ln(XN, G1, conv_ln2_g + iD, conv_ln2_b + iD, nullptr, 1e-5f, 1); // +GELU
    gemm(G1, DM, p_cl + (size_t)i*DM*DM, DM, conv_lin_b + iD, X, X, DM, 0, 1.0f);

    // FFN2 + norm2: X = LN(X + 0.5*ffn2(X))
    ln(X, XN, ffn2_ln_g + iD, ffn2_ln_b + iD, nullptr, 1e-5f, 0);
    gemm(XN, DM, p_f21 + (size_t)i*DF*DM, DM, ffn2_b1 + (size_t)i*DF,
         nullptr, H1, DF, 1, 1.0f);
    gemm(H1, DF, p_f22 + (size_t)i*DM*DF, DF, ffn2_b2 + iD, X, X, DM, 0, 0.5f);
    ln(X, X, norm2_g + iD, norm2_b + iD, nullptr, 1e-5f, 0);
  }

  // ---- final LN (eps=1e-6) + speaker add -> d_out ----
  ln(X, (float*)d_out, final_g, final_b, speaker, 1e-6f, 0);
}
